// CMDPEncoder_45612552683567
// MI455X (gfx1250) — matrix operand order repaired and checked by reading, not compiled, not run on hardware
//
#include <hip/hip_runtime.h>
#include <hip/hip_bf16.h>

// ---------------------------------------------------------------------------
// Problem constants (from the reference)
// ---------------------------------------------------------------------------
#define BB 32
#define SS 512
#define DD 768
#define VV 8192
#define KK 8
#define NTOK (BB * SS)            // 16384 tokens
#define KCH  (DD / 32)            // 24 wmma K-steps of 32
#define NWAVE 8                   // waves per block (256 threads, wave32)
#define NSPLIT 4                  // N (codebook) split across blockIdx.y
#define NCOLS (VV / NSPLIT)       // 2048 columns per block
#define NROWS 32                  // codebook rows staged per chunk
#define NC (NCOLS / NROWS)        // 64 chunks per block
#define ROWBLK (DD * 2 / 16)      // 96 sixteen-byte blocks per codebook row

typedef __bf16 v16bf __attribute__((ext_vector_type(16)));
typedef __bf16 v8bf  __attribute__((ext_vector_type(8)));
typedef float  v8f   __attribute__((ext_vector_type(8)));

union V16 { v16bf v; v8bf p[2]; };

// ---------------------------------------------------------------------------
// Phase 0: codebook f32 -> bf16, plus f32 row norms ||c_v||^2
// ---------------------------------------------------------------------------
__global__ __launch_bounds__(256) void cvt_cb_kernel(const float* __restrict__ cb,
                                                     __bf16* __restrict__ out,
                                                     float* __restrict__ cbsq) {
    const int v = blockIdx.x;
    const int tid = threadIdx.x;
    __shared__ float red[256];
    float acc = 0.0f;
#pragma unroll
    for (int e = 0; e < 3; ++e) {
        int d = tid + 256 * e;
        float x = cb[(size_t)v * DD + d];
        out[(size_t)v * DD + d] = (__bf16)x;
        acc += x * x;
    }
    red[tid] = acc;
    __syncthreads();
    for (int st = 128; st > 0; st >>= 1) {
        if (tid < st) red[tid] += red[tid + st];
        __syncthreads();
    }
    if (tid == 0) cbsq[v] = red[0];
}

// ---------------------------------------------------------------------------
// Phase 1: VQ argmin via bf16 WMMA with LDS-shared, async-staged B tiles.
// Block = 8 waves; each wave owns 16 tokens (M=128/block), all waves share
// double-buffered 32-row codebook chunks staged with
// global_load_async_to_lds_b128 (ASYNCcnt) and XOR-swizzled against LDS bank
// conflicts. The inner K-loop keeps a two-deep B-fragment pipeline so the
// ds_load latency is hidden behind the matrix pipe (counted DScnt waits).
// ---------------------------------------------------------------------------
__global__ __launch_bounds__(256) void vq_argmin_kernel(const float* __restrict__ emb,
                                                        const __bf16* __restrict__ cbB,
                                                        const float* __restrict__ cbsq,
                                                        float* __restrict__ scoreWS,
                                                        int* __restrict__ idxWS) {
    __shared__ __align__(16) __bf16 Bs[2][NROWS * DD];   // 2 x 48 KB

    const int tid  = threadIdx.x;
    const int wave = tid >> 5;
    const int lane = tid & 31;
    const int split = blockIdx.y;
    const int nbase = split * NCOLS;
    const int mwave = blockIdx.x * (16 * NWAVE) + wave * 16;

    // ---- A fragments: convert this wave's 16 token rows f32 -> bf16 once.
    //      (16-bit A 16x32 layout: lane%16 = M; low half K {0..7,16..23},
    //       high half K {8..15,24..31})
    const int rloc = lane & 15;
    const int koff = (lane >> 4) ? 8 : 0;
    const int jsel = (lane >> 4) ? 1 : 0;
    const float* arow = emb + (size_t)(mwave + rloc) * DD;
    v16bf afrag[KCH];
#pragma unroll
    for (int k = 0; k < KCH; ++k) {
        const float* s0 = arow + k * 32 + koff;
        V16 t;
#pragma unroll
        for (int e = 0; e < 8; ++e) t.v[e] = (__bf16)s0[e];
#pragma unroll
        for (int e = 0; e < 8; ++e) t.v[8 + e] = (__bf16)s0[16 + e];
        afrag[k] = t.v;
    }

    float bestS[8];
    int   bestI[8];
#pragma unroll
    for (int r = 0; r < 8; ++r) { bestS[r] = 3.0e38f; bestI[r] = 0; }

    // ---- async stage of one 32-row chunk into buffer `buf` (swizzled)
    auto stage = [&](int buf, int colbase) {
        __bf16* bufp = &Bs[buf][0];
        const char* gbase = (const char*)(cbB + (size_t)colbase * DD);
#pragma unroll
        for (int i = 0; i < (NROWS * ROWBLK) / 256; ++i) {   // 12 blocks/thread
            int o = tid + 256 * i;
            int r = o / ROWBLK;
            int j = o - r * ROWBLK;
            int js = (j & ~15) | ((j ^ r) & 15);             // bank swizzle
            unsigned laddr =
                (unsigned)(size_t)(bufp + (size_t)r * DD + (size_t)js * 8);
            unsigned long long gaddr =
                (unsigned long long)(size_t)(gbase + (size_t)r * (DD * 2) + (size_t)j * 16);
            asm volatile("global_load_async_to_lds_b128 %0, %1, off"
                         :: "v"(laddr), "v"(gaddr) : "memory");
        }
    };

    // ---- prologue: stage chunk 0, drain ASYNCcnt, publish to block
    stage(0, nbase);
    asm volatile("s_wait_asynccnt 0x0" ::: "memory");
    __syncthreads();

    for (int c = 0; c < NC; ++c) {
        if (c + 1 < NC) stage((c + 1) & 1, nbase + (c + 1) * NROWS);

        const __bf16* bs = &Bs[c & 1][0];
#pragma unroll
        for (int t2 = 0; t2 < 2; ++t2) {                 // two 16-col tiles
            const __bf16* bsrow = bs + (t2 * 16 + rloc) * DD;

            // swizzled B-fragment fetch for K-step k into V16
            auto loadB = [&](V16& b, int k) {
                const int j0 = 4 * k + jsel;
            const int j1 = j0 + 2;
                b.p[0] = *(const v8bf*)(bsrow +
                         (((j0 & ~15) | ((j0 ^ rloc) & 15)) << 3));
                b.p[1] = *(const v8bf*)(bsrow +
                         (((j1 & ~15) | ((j1 ^ rloc) & 15)) << 3));
            };

            v8f cacc = {};
            V16 b0, b1;
            loadB(b0, 0);
            loadB(b1, 1);
#pragma unroll
            for (int k = 0; k < KCH; k += 2) {
                cacc = __builtin_amdgcn_wmma_f32_16x16x32_bf16(
                           false, afrag[k], false, b0.v,
                           (short)0, cacc, false, false);
                if (k + 2 < KCH) loadB(b0, k + 2);
                cacc = __builtin_amdgcn_wmma_f32_16x16x32_bf16(
                           false, afrag[k + 1], false, b1.v,
                           (short)0, cacc, false, false);
                if (k + 3 < KCH) loadB(b1, k + 3);
            }

            const int n = nbase + c * NROWS + t2 * 16 + rloc;
            const float cbs = cbsq[n];
#pragma unroll
            for (int r = 0; r < 8; ++r) {
                float s = cbs - 2.0f * cacc[r];          // ||c||^2 - 2 x.c
                if (s < bestS[r]) { bestS[r] = s; bestI[r] = n; }
            }
        }

        if (c + 1 < NC) {
            asm volatile("s_wait_asynccnt 0x0" ::: "memory");
        }
        __syncthreads();
    }

    // ---- per-token argmin over the 16 column-lanes of each half-wave,
    //      then write this split's candidate to workspace
#pragma unroll
    for (int r = 0; r < 8; ++r) {
        float s = bestS[r];
        int   ix = bestI[r];
#pragma unroll
        for (int m = 8; m >= 1; m >>= 1) {
            float os = __shfl_xor(s, m, 32);
            int   oi = __shfl_xor(ix, m, 32);
            if (os < s) { s = os; ix = oi; }
        }
        if ((lane & 15) == 0) {
            int tok = mwave + r + ((lane >> 4) << 3);
            scoreWS[(size_t)split * NTOK + tok] = s;
            idxWS[(size_t)split * NTOK + tok] = ix;
        }
    }
}

// ---------------------------------------------------------------------------
// Phase 1b: reduce the NSPLIT partial argmins per token
// ---------------------------------------------------------------------------
__global__ __launch_bounds__(256) void reduce_splits_kernel(const float* __restrict__ sWS,
                                                            const int* __restrict__ iWS,
                                                            int* __restrict__ idxOut) {
    int t = blockIdx.x * 256 + threadIdx.x;
    if (t < NTOK) {
        float s = sWS[t];
        int   ix = iWS[t];
#pragma unroll
        for (int w = 1; w < NSPLIT; ++w) {
            float os = sWS[(size_t)w * NTOK + t];
            int   oi = iWS[(size_t)w * NTOK + t];
            if (os < s) { s = os; ix = oi; }
        }
        idxOut[t] = ix;
    }
}

// ---------------------------------------------------------------------------
// Phase 2: cross-batch mixing + DP noise (gather-mean over L2-resident rows)
// ---------------------------------------------------------------------------
__global__ __launch_bounds__(256) void mix_noise_kernel(const float* __restrict__ cb,
                                                        const float* __restrict__ noise,
                                                        const int* __restrict__ ridx,
                                                        const int* __restrict__ idx,
                                                        float* __restrict__ out) {
    const int t = blockIdx.x;          // token = b*S + s
    const int s = t % SS;
    const int tid = threadIdx.x;

    int rows[KK];
#pragma unroll
    for (int j = 0; j < KK; ++j) {
        int rb = ridx[t * KK + j];     // random batch index
        rows[j] = idx[rb * SS + s];    // idx[rand_idx[b,s,j], s]
    }

#pragma unroll
    for (int e = 0; e < 3; ++e) {
        int d = tid + 256 * e;
        float acc = 0.0f;
#pragma unroll
        for (int j = 0; j < KK; ++j)
            acc += cb[(size_t)rows[j] * DD + d];
        out[(size_t)t * DD + d] = acc * (1.0f / (float)KK)
                                + 0.1f * noise[(size_t)t * DD + d];
    }
}

// ---------------------------------------------------------------------------
// Launch
// ---------------------------------------------------------------------------
extern "C" void kernel_launch(void* const* d_in, const int* in_sizes, int n_in,
                              void* d_out, int out_size, void* d_ws, size_t ws_size,
                              hipStream_t stream) {
    (void)in_sizes; (void)n_in; (void)out_size; (void)ws_size;

    const float* emb   = (const float*)d_in[0];   // [B,S,D] f32
    const float* cb    = (const float*)d_in[1];   // [V,D]   f32
    const float* noise = (const float*)d_in[2];   // [B,S,D] f32
    const int*   ridx  = (const int*)d_in[3];     // [B,S,K] i32
    float* out = (float*)d_out;                   // [B,S,D] f32

    // workspace layout (bytes)
    char* ws = (char*)d_ws;
    const size_t cbB_bytes   = (size_t)VV * DD * 2;          // 12,582,912
    const size_t cbsq_bytes  = (size_t)VV * 4;               // 32,768
    const size_t part_bytes  = (size_t)NSPLIT * NTOK * 4;    // 262,144 each
    __bf16* cbB     = (__bf16*)ws;
    float*  cbsq    = (float*)(ws + cbB_bytes);
    float*  scoreWS = (float*)(ws + cbB_bytes + cbsq_bytes);
    int*    idxPart = (int*)(ws + cbB_bytes + cbsq_bytes + part_bytes);
    int*    idxWS   = (int*)(ws + cbB_bytes + cbsq_bytes + 2 * part_bytes);

    // Phase 0: codebook -> bf16 + row norms
    cvt_cb_kernel<<<VV, 256, 0, stream>>>(cb, cbB, cbsq);

    // Phase 1: WMMA argmin (grid: 128 M-blocks x 4 N-splits)
    vq_argmin_kernel<<<dim3(NTOK / (16 * NWAVE), NSPLIT), 256, 0, stream>>>(
        emb, cbB, cbsq, scoreWS, idxPart);

    // Phase 1b: fold the N-splits
    reduce_splits_kernel<<<(NTOK + 255) / 256, 256, 0, stream>>>(scoreWS, idxPart, idxWS);

    // Phase 2: gather-mean + noise
    mix_noise_kernel<<<NTOK, 256, 0, stream>>>(cb, noise, ridx, idxWS, out);
}
